// Actor_12738873000286
// MI455X (gfx1250) — matrix operand order repaired and checked by reading, not compiled, not run on hardware
//
#include <hip/hip_runtime.h>
#include <math.h>

typedef _Float16 half16 __attribute__((ext_vector_type(16)));
typedef _Float16 half8  __attribute__((ext_vector_type(8)));
typedef float    float8 __attribute__((ext_vector_type(8)));
typedef float    float4v __attribute__((ext_vector_type(4)));

namespace {
constexpr int   kE            = 128;
constexpr int   kN            = 1024;
constexpr int   kKvStride     = 2 * kE + 3;   // 259 floats per kv row
constexpr int   kRowsPerBlock = 64;           // 4 waves * 16 rows each
constexpr int   kThreads      = 128;          // 4 wave32s
constexpr int   kLdsStride    = kE + 8;       // 136 halves = 272B rows (16B aligned, bank-conflict-dodging)
constexpr float kInvSqrtE     = 0.08838834764831845f; // 1/sqrt(128)
constexpr float kActionScale  = 0.01f;
constexpr float kEps          = 1e-8f;
}

__global__ __launch_bounds__(kThreads)
void actor_fused_kernel(const float* __restrict__ kv,
                        const float* __restrict__ pos,
                        float* __restrict__ out)
{
  // double-buffered f16 V tile: 2 x 16 rows x 136 halves = 8.7 KB
  __shared__ _Float16 ldsB[2][16 * kLdsStride];
  // whole batch's positions: 1024 x 3 floats = 12 KB
  __shared__ float ldsPos[kN * 3];

  const int b    = blockIdx.y;
  const int tid  = threadIdx.x;
  const int wave = tid >> 5;          // wave32
  const int lane = tid & 31;
  const int lo   = lane & 15;
  const int hi   = lane >> 4;
  const int nbase = blockIdx.x * kRowsPerBlock + wave * 16;

  const size_t kvb  = (size_t)b * kN * kKvStride;
  const size_t posb = (size_t)b * kN * 3;

  // ---- stage one 16-row v_mu tile into LDS (f32 -> 4xf16 b64 stores) ----
  auto stage = [&](int buf, int mbase) {
    #pragma unroll
    for (int j = 0; j < (16 * kE / 4) / kThreads; ++j) {  // 4 b64 stores / thread
      const int p   = j * kThreads + tid;                 // quad index, coalesced
      const int row = p >> 5;                             // 32 quads per row
      const int e   = (p & 31) * 4;
      const float* src = kv + kvb + (size_t)(mbase + row) * kKvStride + kE + e;
      union { unsigned long long u; _Float16 h[4]; } pk;
      pk.h[0] = (_Float16)src[0];
      pk.h[1] = (_Float16)src[1];
      pk.h[2] = (_Float16)src[2];
      pk.h[3] = (_Float16)src[3];
      // byte addr = row*272 + (e/4)*8 -> 8B aligned
      *reinterpret_cast<unsigned long long*>(&ldsB[buf][row * kLdsStride + e]) = pk.u;
    }
  };

  // ---- stage batch positions into LDS (16B-aligned float4 copies) ----
  {
    const float4v* src = reinterpret_cast<const float4v*>(pos + posb); // posb*4B % 16 == 0
    float4v* dst = reinterpret_cast<float4v*>(ldsPos);
    #pragma unroll
    for (int j = 0; j < (kN * 3 / 4) / kThreads; ++j)     // 6 per thread
      dst[j * kThreads + tid] = src[j * kThreads + tid];
  }

  stage(0, 0);   // prime the V-tile pipeline

  // ---- A fragments: k_mu rows (f32 -> f16), CDNA5 16-bit A 16x32 layout ----
  // lane (hi,lo): row M = lo; VGPR0-3 hold K = hi*8 + 0..7, VGPR4-7 hold K = hi*8 + 16..23
  half16 afrag[4];
  {
    const float* krow = kv + kvb + (size_t)(nbase + lo) * kKvStride; // k_mu at col 0
    #pragma unroll
    for (int kc = 0; kc < 4; ++kc) {
      union { half16 v; _Float16 h[16]; } u;
      const int base = kc * 32 + hi * 8;
      #pragma unroll
      for (int j = 0; j < 8; ++j) {
        u.h[j]     = (_Float16)krow[base + j];
        u.h[8 + j] = (_Float16)krow[base + 16 + j];
      }
      afrag[kc] = u.v;
    }
  }

  // ---- positions for this lane's C/D rows (M = r + 8*hi), one-time global reads ----
  float pnx[8], pny[8], pnz[8];
  #pragma unroll
  for (int r = 0; r < 8; ++r) {
    const float* p = pos + posb + (size_t)(nbase + r + 8 * hi) * 3;
    pnx[r] = p[0]; pny[r] = p[1]; pnz[r] = p[2];
  }

  float accx[8], accy[8], accz[8];
  #pragma unroll
  for (int r = 0; r < 8; ++r) { accx[r] = 0.f; accy[r] = 0.f; accz[r] = 0.f; }

  for (int mt = 0; mt < kN / 16; ++mt) {
    const int cur   = mt & 1;
    const int mbase = mt * 16;

    __syncthreads();   // tile mt (and, for mt==0, ldsPos) staged; old reads done

    if (mt + 1 < kN / 16) stage(cur ^ 1, mbase + 16);   // overlaps with WMMA below

    // ---- preload all 4 B fragments, then chain 4 WMMAs (pipelined dscnt waits) ----
    // B 32x16 layout: lane (hi,lo) holds column N = lo, K = hi*8 + 0..7 and hi*8 + 16..23
    half16 bfrag[4];
    #pragma unroll
    for (int kc = 0; kc < 4; ++kc) {
      union { half16 v; half8 p2[2]; } ub;
      const half8* bp = (const half8*)&ldsB[cur][lo * kLdsStride + kc * 32 + hi * 8];
      ub.p2[0] = bp[0];
      ub.p2[1] = bp[2];
      bfrag[kc] = ub.v;
    }
    float8 c = {0.f, 0.f, 0.f, 0.f, 0.f, 0.f, 0.f, 0.f};
    #pragma unroll
    for (int kc = 0; kc < 4; ++kc)
      c = __builtin_amdgcn_wmma_f32_16x16x32_f16(false, afrag[kc], false, bfrag[kc],
                                                 (short)0, c, false, false);

    // ---- fused epilogue: acc_d += Pn_d * S  (lane's m = mbase + lo) ----
    const float* pm = &ldsPos[(mbase + lo) * 3];   // ds_load, no HBM-latency stall
    const float pmx = pm[0], pmy = pm[1], pmz = pm[2];
    #pragma unroll
    for (int r = 0; r < 8; ++r) {
      const float dx = pnx[r] - pmx;
      const float dy = pny[r] - pmy;
      const float dz = pnz[r] - pmz;
      const float s2  = dx * dx + dy * dy + dz * dz;
      // 1-ulp hw sqrt/rcp; +eps keeps n==m diagonal finite (dx=0 -> contribution 0)
      const float inv = __builtin_amdgcn_rcpf(__builtin_amdgcn_sqrtf(s2) + kEps);
      const float w   = c[r] * inv;
      accx[r] += dx * w;
      accy[r] += dy * w;
      accz[r] += dz * w;
    }
  }

  // ---- butterfly reduction over the 16 m-lanes in each half-wave ----
  #pragma unroll
  for (int r = 0; r < 8; ++r) {
    #pragma unroll
    for (int mask = 1; mask < 16; mask <<= 1) {
      accx[r] += __shfl_xor(accx[r], mask, 32);
      accy[r] += __shfl_xor(accy[r], mask, 32);
      accz[r] += __shfl_xor(accz[r], mask, 32);
    }
  }

  // lane (hi, lo=r) writes row nbase + r + 8*hi
  if (lo < 8) {
    const int r = lo;
    const int n = nbase + r + 8 * hi;
    float* o = out + posb + (size_t)n * 3;
    o[0] = kActionScale * tanhf(accx[r] * kInvSqrtE);
    o[1] = kActionScale * tanhf(accy[r] * kInvSqrtE);
    o[2] = kActionScale * tanhf(accz[r] * kInvSqrtE);
  }
}

extern "C" void kernel_launch(void* const* d_in, const int* in_sizes, int n_in,
                              void* d_out, int out_size, void* d_ws, size_t ws_size,
                              hipStream_t stream)
{
  const float* kv  = (const float*)d_in[0];
  const float* pos = (const float*)d_in[1];
  float* out = (float*)d_out;
  const int B = in_sizes[0] / (kN * kKvStride);   // 16
  dim3 grid(kN / kRowsPerBlock, B, 1);
  actor_fused_kernel<<<grid, dim3(kThreads, 1, 1), 0, stream>>>(kv, pos, out);
  (void)n_in; (void)out_size; (void)d_ws; (void)ws_size;
}
